// MoE_20753281974372
// MI455X (gfx1250) — matrix operand order repaired and checked by reading, not compiled, not run on hardware
//
#include <hip/hip_runtime.h>
#include <hip/hip_bf16.h>

typedef __bf16 bf16_t;
typedef __attribute__((ext_vector_type(16))) __bf16 v16bf;
typedef __attribute__((ext_vector_type(8)))  __bf16 v8bf;
typedef __attribute__((ext_vector_type(8)))  float  v8f;
typedef __attribute__((ext_vector_type(4)))  unsigned int v4u;
typedef __attribute__((ext_vector_type(8)))  unsigned int v8u;

#define D_MODEL 2048
#define D_INT   4096
#define N_EXP   8
#define T_TOK   8192          // BATCH * SEQ

#define TILE_M 128
#define TILE_N 128
#define TILE_K 32
#define LDS_STRIDE 40         // bf16 per row: 32 data + 8 pad (80B, 16B aligned)

// ---------------------------------------------------------------- helpers

__device__ __forceinline__ v16bf combine16(v8bf lo, v8bf hi) {
    v16bf r;
#pragma unroll
    for (int i = 0; i < 8; ++i) { r[i] = lo[i]; r[i + 8] = hi[i]; }
    return r;
}

// A fragment, 16x32 bf16 (CDNA5 ISA layout): lane l holds row rbase+(l&15);
// elements j = K[lh*16 + j] for j=0..15 (contiguous along K).
__device__ __forceinline__ v16bf load_a_frag(const bf16_t* s, int rbase, int lane) {
    int r  = rbase + (lane & 15);
    int kb = (lane >> 4) * 16;
    v8bf lo = *(const v8bf*)(s + r * LDS_STRIDE + kb);
    v8bf hi = *(const v8bf*)(s + r * LDS_STRIDE + kb + 8);
    return combine16(lo, hi);
}

// B fragment, 32x16 bf16: lane l holds col nbase+(l&15);
// elements j = K[lh*16 + j] for j=0..15 (contiguous along K of the weight row).
__device__ __forceinline__ v16bf load_b_frag(const bf16_t* s, int nbase, int lane) {
    int n  = nbase + (lane & 15);
    int k0 = (lane >> 4) * 16;
    v8bf lo = *(const v8bf*)(s + n * LDS_STRIDE + k0);
    v8bf hi = *(const v8bf*)(s + n * LDS_STRIDE + k0 + 8);
    return combine16(lo, hi);
}

// load 16 fp32 from global (or zeros), convert to bf16, store 32B to LDS
__device__ __forceinline__ void stage_f32(const float* __restrict__ src,
                                          bf16_t* dst, bool valid) {
    float t[16];
    if (valid) {
        const float4* p = (const float4*)src;
#pragma unroll
        for (int j = 0; j < 4; ++j) ((float4*)t)[j] = p[j];
    } else {
#pragma unroll
        for (int j = 0; j < 16; ++j) t[j] = 0.f;
    }
    v8bf a, b;
#pragma unroll
    for (int j = 0; j < 8; ++j) { a[j] = (bf16_t)t[j]; b[j] = (bf16_t)t[j + 8]; }
    *(v8bf*)dst       = a;
    *(v8bf*)(dst + 8) = b;
}

// ---------------------------------------------------------------- router

__global__ __launch_bounds__(256) void moe_router(const float* __restrict__ x,
                                                  const float* __restrict__ Wr,
                                                  int*   __restrict__ tok_e,
                                                  float* __restrict__ tok_w) {
    int wave = threadIdx.x >> 5;
    int lane = threadIdx.x & 31;
    int t = blockIdx.x * 8 + wave;
    if (t >= T_TOK) return;
    const float* xt = x + (size_t)t * D_MODEL;

    float acc[N_EXP];
#pragma unroll
    for (int e = 0; e < N_EXP; ++e) acc[e] = 0.f;

    for (int d = lane; d < D_MODEL; d += 32) {
        float xv = xt[d];
#pragma unroll
        for (int e = 0; e < N_EXP; ++e) acc[e] += xv * Wr[e * D_MODEL + d];
    }
#pragma unroll
    for (int e = 0; e < N_EXP; ++e) {
#pragma unroll
        for (int off = 16; off > 0; off >>= 1)
            acc[e] += __shfl_xor(acc[e], off, 32);
    }
    if (lane == 0) {
        int e0 = 0;
#pragma unroll
        for (int e = 1; e < N_EXP; ++e) if (acc[e] > acc[e0]) e0 = e;
        int e1 = (e0 == 0) ? 1 : 0;
#pragma unroll
        for (int e = 0; e < N_EXP; ++e)
            if (e != e0 && acc[e] > acc[e1]) e1 = e;
        float w0 = 1.f / (1.f + __expf(acc[e1] - acc[e0]));  // softmax over top-2
        tok_e[2 * t]     = e0;  tok_e[2 * t + 1] = e1;
        tok_w[2 * t]     = w0;  tok_w[2 * t + 1] = 1.f - w0;
    }
}

// ------------------------------------------------- deterministic compaction

__global__ __launch_bounds__(32) void moe_build_lists(const int*   __restrict__ tok_e,
                                                      const float* __restrict__ tok_w,
                                                      int*   __restrict__ idx,
                                                      float* __restrict__ wgt,
                                                      int*   __restrict__ cnt) {
    int e = blockIdx.x;
    int lane = threadIdx.x;
    int count = 0;
    for (int base = 0; base < T_TOK; base += 32) {
        int t   = base + lane;
        int a0  = tok_e[2 * t], a1 = tok_e[2 * t + 1];
        bool sel = (a0 == e) || (a1 == e);
        float w  = (a0 == e) ? tok_w[2 * t] : tok_w[2 * t + 1];
        unsigned long long m = __ballot(sel);
        if (sel) {
            int pos = count + __popcll(m & ((1ull << lane) - 1ull));
            idx[(size_t)e * T_TOK + pos] = t;
            wgt[(size_t)e * T_TOK + pos] = w;
        }
        count += __popcll(m);
    }
    if (lane == 0) cnt[e] = count;
}

__global__ void moe_zero(float* __restrict__ y, int n) {
    int i = blockIdx.x * blockDim.x + threadIdx.x;
    if (i < n) y[i] = 0.f;
}

// ------------------------------------------------- GEMM1: H = silu(Xg @ Win^T)

__global__ __launch_bounds__(256) void moe_gemm1(const float* __restrict__ x,
                                                 const float* __restrict__ Win,
                                                 const int*   __restrict__ idx,
                                                 const int*   __restrict__ cnt,
                                                 bf16_t*      __restrict__ H) {
    __shared__ bf16_t As[TILE_M * LDS_STRIDE];
    __shared__ bf16_t Bs[TILE_N * LDS_STRIDE];

    int e  = blockIdx.y;
    int ce = cnt[e];
    int m0 = blockIdx.x * TILE_M;
    if (m0 >= ce) return;                    // block-uniform early exit
    int nt = blockIdx.z;

    int offs = 0;                            // row offset of this expert in H
    for (int i = 0; i < e; ++i) offs += cnt[i];

    int tid  = threadIdx.x;
    int rowL = tid >> 1;                     // 0..127: row this thread stages
    int half = tid & 1;                      // which 16-wide K half
    bool avalid = (m0 + rowL) < ce;
    int  tok    = avalid ? idx[(size_t)e * T_TOK + m0 + rowL] : 0;
    const float* asrc = x + (size_t)tok * D_MODEL + half * 16;
    const float* bsrc = Win + ((size_t)e * D_INT + nt * TILE_N + rowL) * D_MODEL
                        + half * 16;
    bf16_t* adst = As + rowL * LDS_STRIDE + half * 16;
    bf16_t* bdst = Bs + rowL * LDS_STRIDE + half * 16;

    int wave = tid >> 5, lane = tid & 31;
    int wm = wave >> 2, wn = wave & 3;       // 2x4 wave grid -> 64x32 per wave

    v8f acc[4][2] = {};

    for (int k0 = 0; k0 < D_MODEL; k0 += TILE_K) {
        if (k0 + TILE_K < D_MODEL) {
            __builtin_prefetch(asrc + k0 + TILE_K, 0, 3);
            __builtin_prefetch(bsrc + k0 + TILE_K, 0, 3);
        }
        stage_f32(asrc + k0, adst, avalid);
        stage_f32(bsrc + k0, bdst, true);
        __syncthreads();

        v16bf a[4], b[2];
#pragma unroll
        for (int rm = 0; rm < 4; ++rm) a[rm] = load_a_frag(As, wm * 64 + rm * 16, lane);
#pragma unroll
        for (int cn = 0; cn < 2; ++cn) b[cn] = load_b_frag(Bs, wn * 32 + cn * 16, lane);
#pragma unroll
        for (int rm = 0; rm < 4; ++rm)
#pragma unroll
            for (int cn = 0; cn < 2; ++cn)
                acc[rm][cn] = __builtin_amdgcn_wmma_f32_16x16x32_bf16(
                    false, a[rm], false, b[cn], (short)0, acc[rm][cn], false, false);
        __syncthreads();
    }

    int lr = lane & 15, lh = lane >> 4;
#pragma unroll
    for (int rm = 0; rm < 4; ++rm)
#pragma unroll
        for (int cn = 0; cn < 2; ++cn)
#pragma unroll
            for (int r = 0; r < 8; ++r) {
                int row = wm * 64 + rm * 16 + lh * 8 + r;
                if (m0 + row < ce) {
                    float v = acc[rm][cn][r];
                    float s = v / (1.f + __expf(-v));      // SiLU
                    int col = nt * TILE_N + wn * 32 + cn * 16 + lr;
                    H[(size_t)(offs + m0 + row) * D_INT + col] = (bf16_t)s;
                }
            }
}

// ------------------------------------------------- GEMM2: y += w * (H @ Wout^T)
// A tile (bf16 H rows) fetched by the Tensor Data Mover straight into LDS,
// with D# padding reproducing the 80B conflict-free row stride and
// tensor_dim1 OOB-zeroing rows past cnt[e].

__global__ __launch_bounds__(256) void moe_gemm2(const bf16_t* __restrict__ H,
                                                 const float*  __restrict__ Wout,
                                                 const int*    __restrict__ idx,
                                                 const float*  __restrict__ wgt,
                                                 const int*    __restrict__ cnt,
                                                 float*        __restrict__ y) {
    __shared__ bf16_t As[TILE_M * LDS_STRIDE];
    __shared__ bf16_t Bs[TILE_N * LDS_STRIDE];

    int e  = blockIdx.y;
    int ce = cnt[e];
    int m0 = blockIdx.x * TILE_M;
    if (m0 >= ce) return;
    int nt = blockIdx.z;

    int offs = 0;
    for (int i = 0; i < e; ++i) offs += cnt[i];

    int rem = ce - m0;                        // valid rows in this tile
    if (rem > TILE_M) rem = TILE_M;

    int tid  = threadIdx.x;
    int rowL = tid >> 1;
    int half = tid & 1;
    const float* bsrc = Wout + ((size_t)e * D_MODEL + nt * TILE_N + rowL) * D_INT
                        + half * 16;
    bf16_t* bdst = Bs + rowL * LDS_STRIDE + half * 16;

    int wave = tid >> 5, lane = tid & 31;
    int wm = wave >> 2, wn = wave & 3;

    // ---- Tensor DMA descriptor (uniform; group0 global_addr updated per k0)
    unsigned int       lds_base = (unsigned int)(size_t)(void*)As;  // LDS byte offset
    unsigned long long gbase    = (unsigned long long)(size_t)(const void*)H
                                + (unsigned long long)(offs + m0) * D_INT * 2ull;
    v8u g1;
    g1[0] = (1u << 16)                 // data_size = 2 bytes
          | (1u << 20)                 // pad_enable
          | (3u << 22)                 // pad_interval: every 16 DWORDs (64B row)
          | (3u << 25);                // pad_amount: 4 DWORDs (16B) -> 80B stride
    g1[1] = (32u & 0xFFFFu) << 16;     // tensor_dim0 = 32 (bits 79:48, low half)
    g1[2] = ((unsigned)rem & 0xFFFFu) << 16;              // tensor_dim1 lo16
    g1[3] = (((unsigned)rem >> 16) & 0xFFFFu)             // tensor_dim1 hi16
          | (32u << 16);                                  // tile_dim0 = 32
    g1[4] = (unsigned)TILE_M;          // tile_dim1 = 128, tile_dim2 = 0
    g1[5] = (unsigned)D_INT;           // tensor_dim0_stride = 4096 (lo32)
    g1[6] = 0u;                        // stride hi16 | tensor_dim1_stride lo16
    g1[7] = 0u;

    v8f acc[4][2] = {};

    for (int k0 = 0; k0 < D_INT; k0 += TILE_K) {
        if (wave == 0) {                               // one wave drives the TDM
            unsigned long long ga = gbase + (unsigned long long)k0 * 2ull;
            v4u g0;
            g0[0] = 1u;                                // count = 1, user mode
            g0[1] = lds_base;                          // LDS tile base
            g0[2] = (unsigned int)(ga & 0xFFFFFFFFull);
            g0[3] = ((unsigned int)(ga >> 32) & 0x01FFFFFFu) | (2u << 30); // type=2
            asm volatile("tensor_load_to_lds %0, %1"
                         :: "s"(g0), "s"(g1) : "memory");
        }
        if (k0 + TILE_K < D_INT)
            __builtin_prefetch(bsrc + k0 + TILE_K, 0, 3);
        stage_f32(bsrc + k0, bdst, true);              // Wout fp32 -> bf16
        if (wave == 0)
            __builtin_amdgcn_s_wait_tensorcnt(0);      // A tile resident in LDS
        __syncthreads();                               // publish A (TDM) + B (ds)

        v16bf a[4], b[2];
#pragma unroll
        for (int rm = 0; rm < 4; ++rm) a[rm] = load_a_frag(As, wm * 64 + rm * 16, lane);
#pragma unroll
        for (int cn = 0; cn < 2; ++cn) b[cn] = load_b_frag(Bs, wn * 32 + cn * 16, lane);
#pragma unroll
        for (int rm = 0; rm < 4; ++rm)
#pragma unroll
            for (int cn = 0; cn < 2; ++cn)
                acc[rm][cn] = __builtin_amdgcn_wmma_f32_16x16x32_bf16(
                    false, a[rm], false, b[cn], (short)0, acc[rm][cn], false, false);
        __syncthreads();
    }

    int lr = lane & 15, lh = lane >> 4;
#pragma unroll
    for (int rm = 0; rm < 4; ++rm)
#pragma unroll
        for (int cn = 0; cn < 2; ++cn)
#pragma unroll
            for (int r = 0; r < 8; ++r) {
                int row = wm * 64 + rm * 16 + lh * 8 + r;
                int gi  = m0 + row;
                if (gi < ce) {
                    int   t = idx[(size_t)e * T_TOK + gi];
                    float w = wgt[(size_t)e * T_TOK + gi];
                    int col = nt * TILE_N + wn * 32 + cn * 16 + lr;
                    // exactly 2 commutative f32 adds per element onto a zeroed
                    // buffer -> bit-deterministic
                    atomicAdd(&y[(size_t)t * D_MODEL + col], w * acc[rm][cn][r]);
                }
            }
}

// ---------------------------------------------------------------- launch

extern "C" void kernel_launch(void* const* d_in, const int* in_sizes, int n_in,
                              void* d_out, int out_size, void* d_ws, size_t ws_size,
                              hipStream_t stream) {
    const float* x    = (const float*)d_in[0];
    const float* Wr   = (const float*)d_in[1];
    const float* Win  = (const float*)d_in[2];
    const float* Wout = (const float*)d_in[3];
    float* y = (float*)d_out;

    char* ws = (char*)d_ws;
    int*    tok_e = (int*)   (ws);                 //  64 KB
    float*  tok_w = (float*) (ws + (64  << 10));   //  64 KB
    int*    idx   = (int*)   (ws + (128 << 10));   // 256 KB
    float*  wgt   = (float*) (ws + (384 << 10));   // 256 KB
    int*    cnt   = (int*)   (ws + (640 << 10));   //  32 B
    bf16_t* H     = (bf16_t*)(ws + (1   << 20));   // 16384 * 4096 * 2 = 128 MB

    moe_router     <<<T_TOK / 8, 256, 0, stream>>>(x, Wr, tok_e, tok_w);
    moe_build_lists<<<N_EXP, 32, 0, stream>>>(tok_e, tok_w, idx, wgt, cnt);
    moe_zero       <<<(T_TOK * D_MODEL + 255) / 256, 256, 0, stream>>>(y, T_TOK * D_MODEL);

    dim3 g1(T_TOK / TILE_M, N_EXP, D_INT / TILE_N);    // 64 x 8 x 32
    moe_gemm1<<<g1, 256, 0, stream>>>(x, Win, idx, cnt, H);

    dim3 g2(T_TOK / TILE_M, N_EXP, D_MODEL / TILE_N);  // 64 x 8 x 16
    moe_gemm2<<<g2, 256, 0, stream>>>(H, Wout, idx, wgt, cnt, y);
}
